// AQLinear_46832323395777
// MI455X (gfx1250) — compile-verified, matrix operand order checked by reading, not run on hardware
//
#include <hip/hip_runtime.h>
#include <hip/hip_bf16.h>
#include <stdint.h>

// ---------------------------------------------------------------------------
// AQLM-style quantized linear for MI455X (gfx1250, wave32, WMMA).
//   out[64 x 12288] = (x/x_max).f16  @  W^T  * (x_max*cb_max*scales[o])
//   W[o, g*5+j] = cb_safe[0, idx[o,g,0], j] + cb_safe[1, idx[o,g,1], j]
// Fused: W tiles are dequantized into LDS and fed to v_wmma_f32_16x16x32_f16,
// never materialized in HBM. Dominant traffic = indices (80.6 MB), read once,
// batched into b64 load clauses (16 pairs / lane / chunk at imm offsets).
// ---------------------------------------------------------------------------

typedef _Float16 v16h __attribute__((ext_vector_type(16)));
typedef _Float16 v8h  __attribute__((ext_vector_type(8)));
typedef float    v8f  __attribute__((ext_vector_type(8)));

#define T_ROWS   64
#define IN_F     4100
#define OUT_F    12288
#define NGRP     820
#define NCODE    256
#define GRP      5
#define KPAD     4160      // 26 * 160
#define CHUNK_G  32        // groups per chunk
#define CHUNK_K  160       // = CHUNK_G * GRP = 5 wmma k-steps of 32
#define NCHUNK   26
#define BT_STRIDE 168      // 160 halves + 16B pad (keeps 16B alignment, skews banks)

// ---------------- kernel 0: init ws scalars (ws poisoned by harness) --------
__global__ void aq_init_kernel(uint32_t* wsu) {
  if (threadIdx.x == 0) { wsu[0] = 0u; wsu[1] = 0u; }
}

// ---------------- kernel 1: abs-max reductions ------------------------------
__device__ inline float block_reduce_max(float v, float* red) {
  #pragma unroll
  for (int off = 16; off > 0; off >>= 1)
    v = fmaxf(v, __shfl_xor(v, off, 32));
  const int lane = threadIdx.x & 31;
  const int wave = threadIdx.x >> 5;
  __syncthreads();                       // allow helper reuse of `red`
  if (lane == 0) red[wave] = v;
  __syncthreads();
  float r = (threadIdx.x < 8) ? red[threadIdx.x] : 0.0f;
  if (wave == 0) {
    #pragma unroll
    for (int off = 4; off > 0; off >>= 1)
      r = fmaxf(r, __shfl_xor(r, off, 32));
  }
  return r;                              // valid in thread 0
}

__global__ void aq_absmax_kernel(const float* __restrict__ x,
                                 const _Float16* __restrict__ cb,
                                 uint32_t* __restrict__ wsu) {
  __shared__ float red[8];
  const int stride = gridDim.x * blockDim.x;
  int gtid = blockIdx.x * blockDim.x + threadIdx.x;

  float mx = 0.0f;
  for (int i = gtid; i < T_ROWS * IN_F; i += stride)
    mx = fmaxf(mx, fabsf(x[i]));
  mx = block_reduce_max(mx, red);
  if (threadIdx.x == 0) atomicMax((int*)&wsu[0], __float_as_int(mx));

  if (blockIdx.x == 0) {
    float mc = 0.0f;
    for (int i = threadIdx.x; i < 2 * NCODE * GRP; i += blockDim.x)
      mc = fmaxf(mc, fabsf((float)cb[i]));
    mc = block_reduce_max(mc, red);
    if (threadIdx.x == 0) atomicMax((int*)&wsu[1], __float_as_int(mc));
  }
}

// ---------------- kernel 2: fp16 conversion with padding --------------------
// x_safe: [64][4160] halves, zero-padded k >= 4100.
// cb_safe: [2][256][8] halves, row padded 5 -> 8 so dequant is one 16B read.
__global__ void aq_convert_kernel(const float* __restrict__ x,
                                  const _Float16* __restrict__ cb,
                                  const float* __restrict__ wsf,
                                  _Float16* __restrict__ xs,
                                  _Float16* __restrict__ cbo) {
  const float xmax  = fmaxf(wsf[0], 1.0f) * 8.0f;
  const float cbmax = fmaxf(wsf[1], 1.0f);
  const int stride = gridDim.x * blockDim.x;
  int gtid = blockIdx.x * blockDim.x + threadIdx.x;

  for (int i = gtid; i < T_ROWS * KPAD; i += stride) {
    const int row = i / KPAD;
    const int k   = i - row * KPAD;
    float v = (k < IN_F) ? (x[row * IN_F + k] / xmax) : 0.0f;
    xs[i] = (_Float16)v;
  }
  for (int i = gtid; i < 2 * NCODE * 8; i += stride) {
    const int j    = i & 7;
    const int code = (i >> 3) & 255;
    const int cs   = i >> 11;
    float v = (j < GRP) ? ((float)cb[(cs * NCODE + code) * GRP + j] / cbmax)
                        : 0.0f;
    cbo[i] = (_Float16)v;
  }
}

// ---------------- kernel 3: fused dequant + WMMA GEMM -----------------------
// Block: 256 threads = 8 waves; each wave owns 16 output columns, all 64 rows.
// Lane mapping for dequant: with t = lane + 32*it, t&15 == n (constant) and
// t>>4 == hi + 2*it, so each lane owns column n and groups {hi, hi+2, ...}:
// consecutive iterations step +32B in the index array -> one b64 load clause.
__global__ __launch_bounds__(256)
void aq_gemm_kernel(const int* __restrict__ idx,
                    const float* __restrict__ scales,
                    const _Float16* __restrict__ xs,
                    const _Float16* __restrict__ cbws,
                    const float* __restrict__ wsf,
                    float* __restrict__ out) {
  __shared__ _Float16 cbL[2 * NCODE * 8];            // 8 KB
  __shared__ _Float16 btile_all[8 * 16 * BT_STRIDE]; // 43 KB

  const int tid = threadIdx.x;
  for (int t = tid; t < 2 * NCODE * 8; t += 256) cbL[t] = cbws[t];
  __syncthreads();

  const int lane = tid & 31;
  const int wave = tid >> 5;
  const int hi   = lane >> 4;          // 0: lanes 0-15, 1: lanes 16-31
  const int n    = lane & 15;
  const int col_base = blockIdx.x * 128 + wave * 16;
  _Float16* btile = btile_all + wave * 16 * BT_STRIDE;

  // Per-lane bases for the dequant phase.
  const int* ibase0 = idx + (col_base + n) * (NGRP * 2) + hi * 2;
  _Float16*  dst0   = btile + n * BT_STRIDE + hi * GRP;

  v8f acc[4];
  #pragma unroll
  for (int mb = 0; mb < 4; ++mb)
    #pragma unroll
    for (int q = 0; q < 8; ++q) acc[mb][q] = 0.0f;

  for (int c = 0; c < NCHUNK; ++c) {
    const int gbase = c * CHUNK_G;
    const int* ib = ibase0 + gbase * 2;

    if (gbase + CHUNK_G <= NGRP) {
      // ---- fast path: batch-load 16 index pairs (one clause, one wait) ----
      int2 iv[16];
      #pragma unroll
      for (int it = 0; it < 16; ++it)
        iv[it] = *(const int2*)(ib + it * 8);
      // prefetch next chunk's 256B index strip (global_prefetch_b8)
      __builtin_prefetch(ib + CHUNK_G * 2, 0, 1);
      __builtin_prefetch(ib + CHUNK_G * 2 + 32, 0, 1);
      #pragma unroll
      for (int it = 0; it < 16; ++it) {
        v8h a = *(const v8h*)(cbL + iv[it].x * 8);
        v8h b = *(const v8h*)(cbL + 2048 + iv[it].y * 8);
        v8h s = a + b;
        _Float16* dst = dst0 + it * (2 * GRP);
        dst[0] = s[0]; dst[1] = s[1]; dst[2] = s[2]; dst[3] = s[3]; dst[4] = s[4];
      }
    } else {
      // ---- tail chunk: guard g < NGRP, zero-fill padding ----
      #pragma unroll
      for (int it = 0; it < 16; ++it) {
        const int g = gbase + hi + 2 * it;
        v8h s;
        #pragma unroll
        for (int q = 0; q < 8; ++q) s[q] = (_Float16)0.0f;
        if (g < NGRP) {
          const int2 iv = *(const int2*)(ib + it * 8);
          v8h a = *(const v8h*)(cbL + iv.x * 8);
          v8h b = *(const v8h*)(cbL + 2048 + iv.y * 8);
          s = a + b;
        }
        _Float16* dst = dst0 + it * (2 * GRP);
        dst[0] = s[0]; dst[1] = s[1]; dst[2] = s[2]; dst[3] = s[3]; dst[4] = s[4];
      }
    }
    // LDS ops from one wave execute in order; wait so cross-lane tile writes
    // are visible before fragment loads below.
    asm volatile("s_wait_dscnt 0" ::: "memory");

    // ---- 5 WMMA k-steps over this chunk ----
    #pragma unroll
    for (int ks = 0; ks < 5; ++ks) {
      // B fragment 32x16 f16: lane owns column n; K runs {0..7,16..23} for
      // lanes 0-15 and {8..15,24..31} for lanes 16-31 (mirrors A layout).
      const _Float16* bp = btile + n * BT_STRIDE + ks * 32 + hi * 8;
      v16h bfrag;
      ((uint4*)&bfrag)[0] = *(const uint4*)(bp);
      ((uint4*)&bfrag)[1] = *(const uint4*)(bp + 16);

      const int kg = c * CHUNK_K + ks * 32 + hi * 8;
      #pragma unroll
      for (int mb = 0; mb < 4; ++mb) {
        // A fragment 16x32 f16 per documented layout (row = M, two 8-K runs).
        const _Float16* ap = xs + (mb * 16 + n) * KPAD + kg;
        v16h afrag;
        ((uint4*)&afrag)[0] = *(const uint4*)(ap);
        ((uint4*)&afrag)[1] = *(const uint4*)(ap + 16);
        acc[mb] = __builtin_amdgcn_wmma_f32_16x16x32_f16(
            false, afrag, false, bfrag, (short)0, acc[mb], false, false);
      }
    }
  }

  // ---- epilogue: scale, nan_to_num, store (C/D layout: VGPR r -> M) ----
  const float gscale = (fmaxf(wsf[0], 1.0f) * 8.0f) * fmaxf(wsf[1], 1.0f);
  const int col = col_base + n;
  const float sc = gscale * scales[col];
  #pragma unroll
  for (int mb = 0; mb < 4; ++mb) {
    #pragma unroll
    for (int r = 0; r < 8; ++r) {
      const int m = mb * 16 + hi * 8 + r;
      float v = acc[mb][r] * sc;
      if (!(fabsf(v) <= 3.402823466e38f)) v = 0.0f;  // NaN/Inf -> 0
      out[m * OUT_F + col] = v;
    }
  }
}

// ---------------------------------------------------------------------------
extern "C" void kernel_launch(void* const* d_in, const int* in_sizes, int n_in,
                              void* d_out, int out_size, void* d_ws, size_t ws_size,
                              hipStream_t stream) {
  (void)in_sizes; (void)n_in; (void)out_size; (void)ws_size;
  const float*    x      = (const float*)d_in[0];
  const int*      idx    = (const int*)d_in[1];
  const _Float16* cb     = (const _Float16*)d_in[2];
  const float*    scales = (const float*)d_in[3];
  float*          out    = (float*)d_out;

  // ws layout: [0,16) two scalar slots; [16, 16+8192) cb_safe; then x_safe.
  uint32_t*    wsu = (uint32_t*)d_ws;
  const float* wsf = (const float*)d_ws;
  _Float16*    cbo = (_Float16*)((char*)d_ws + 16);
  _Float16*    xsf = (_Float16*)((char*)d_ws + 16 + 8192);

  aq_init_kernel<<<1, 64, 0, stream>>>(wsu);
  aq_absmax_kernel<<<256, 256, 0, stream>>>(x, cb, wsu);
  aq_convert_kernel<<<512, 256, 0, stream>>>(x, cb, wsf, xsf, cbo);
  aq_gemm_kernel<<<OUT_F / 128, 256, 0, stream>>>(idx, scales, xsf, cbo, wsf, out);
}
